// DoubleConv_36910948942429
// MI455X (gfx1250) — matrix operand order repaired and checked by reading, not compiled, not run on hardware
//
#include <hip/hip_runtime.h>
#include <hip/hip_fp16.h>

typedef __attribute__((ext_vector_type(16))) _Float16 v16h;
typedef __attribute__((ext_vector_type(8)))  _Float16 v8h;
typedef __attribute__((ext_vector_type(8)))  float    v8f;

// Fixed problem geometry (powers of two -> all index math is shifts/masks)
constexpr int GH   = 32;        // H
constexpr int GW   = 64;        // W
constexpr int GHW  = GH * GW;   // 2048
constexpr int GF   = 64;        // hidden channels
constexpr int GCO  = 256;       // 4*F gate channels
constexpr int NTIL = GCO / 16;  // 16 n-tiles

// ---------------------------------------------------------------------------
// Utility kernels
// ---------------------------------------------------------------------------
__global__ void k_f32_to_f16(const float* __restrict__ src, _Float16* __restrict__ dst, int n) {
    int i = blockIdx.x * blockDim.x + threadIdx.x;
    if (i < n) dst[i] = (_Float16)src[i];
}

__global__ void k_zero_f32(float* __restrict__ p, int n) {
    int i = blockIdx.x * blockDim.x + threadIdx.x;
    if (i < n) p[i] = 0.f;
}

__global__ void k_zero_f16(_Float16* __restrict__ p, int n) {
    int i = blockIdx.x * blockDim.x + threadIdx.x;
    if (i < n) p[i] = (_Float16)0.f;
}

// Pack two HWIO conv weights (concatenated along K = 9*Cin) into the per-lane
// WMMA B-fragment layout: out[((chunk*NTIL + nt)*32 + lane)*16 + e]
//   lane in [0,32), e in [0,16): K = chunk*32 + (lane>>4)*16 + e, N = nt*16 + (lane&15)
__global__ void k_pack_weights(const float* __restrict__ WA, int KA,
                               const float* __restrict__ WB, int KB,
                               int Kpad, _Float16* __restrict__ out) {
    int idx = blockIdx.x * blockDim.x + threadIdx.x;
    int total = Kpad * GCO;
    if (idx >= total) return;
    int e  = idx & 15;
    int t1 = idx >> 4;
    int l  = t1 & 31;
    int t2 = t1 >> 5;
    int nt = t2 % NTIL;
    int c  = t2 / NTIL;
    int k  = c * 32 + ((l >> 4) << 4) + e;
    int n  = (nt << 4) + (l & 15);
    float v = 0.f;
    if (k < KA)            v = WA[(long)k * GCO + n];
    else if (k < KA + KB)  v = WB[(long)(k - KA) * GCO + n];
    out[idx] = (_Float16)v;
}

// ---------------------------------------------------------------------------
// Fully fused ConvLSTM step:  z = bias + conv3x3(x_t, Wx) + conv3x3(h_prev, Wh)
// -> gates (hard_sigmoid i,f,o; tanh g) -> c,h update -> LayerNorm(C=64) +
// LeakyReLU(0.3) written to the sequence output.
//
// Block = 128 threads (4 waves). Block tile = 32 positions x all 256 gate
// channels; each wave computes 32(M) x 64(N) via 8 WMMA per K-chunk
// (2 A-frags x 4 B-frags). z lives only in LDS (32 KB), never in global.
// h-state is ping-ponged across steps (hprev read-only, hnext write-only)
// to avoid cross-block halo races.
// ---------------------------------------------------------------------------
template<int LCA, int KA, int KTOT, int CHUNKS>
__global__ __launch_bounds__(128)
void k_step_fused(const _Float16* __restrict__ srcA, long strideA,
                  const _Float16* __restrict__ hprev16,      // (B,HW,F) f16
                  const _Float16* __restrict__ Wswz,
                  const float* __restrict__ bias,
                  float* __restrict__ cstate,                // (B*HW, F)
                  _Float16* __restrict__ hnext16,            // (B*HW, F)
                  const float* __restrict__ gamma,
                  const float* __restrict__ beta,
                  _Float16* __restrict__ o16, float* __restrict__ o32,
                  int T, int t)
{
    __shared__ float zbuf[32 * GCO];   // 32 KB

    const int lane = threadIdx.x & 31;
    const int wav  = threadIdx.x >> 5;   // n-group 0..3 (64 channels each)
    const int g    = lane >> 4;
    const int mr   = lane & 15;
    const int mt   = blockIdx.x;         // 32-position slab

    // two A rows per lane (M = mt*32 + mr and +16); all shifts/masks
    const int m0 = mt * 32 + mr;
    const int m1 = m0 + 16;
    const int bi0 = m0 >> 11, r0 = m0 & (GHW - 1), y0 = r0 >> 6, x0 = r0 & (GW - 1);
    const int bi1 = m1 >> 11, r1 = m1 & (GHW - 1), y1 = r1 >> 6, x1 = r1 & (GW - 1);
    const _Float16* __restrict__ pA0 = srcA    + (long)bi0 * strideA;
    const _Float16* __restrict__ pA1 = srcA    + (long)bi1 * strideA;
    const _Float16* __restrict__ pB0 = hprev16 + (long)bi0 * (GHW * GF);
    const _Float16* __restrict__ pB1 = hprev16 + (long)bi1 * (GHW * GF);

    // ---------------- implicit-GEMM conv ----------------
    v8f acc[8] = {};   // [mi*4 + ni]
    for (int kc = 0; kc < CHUNKS; ++kc) {
        // A fragments: 2 rows x 2 runs of 8 contiguous halves (one b128 each).
        // Runs are 8-aligned and never straddle a conv tap (Cin in {8,64}).
        v16h af0, af1;
#pragma unroll
        for (int run = 0; run < 2; ++run) {
            const int kr = kc * 32 + g * 16 + run * 8;
            v8h seg0 = {}; v8h seg1 = {};
            if (kr < KTOT) {
                const _Float16 *sp0, *sp1;
                int lc, ks;
                if (kr < KA) { sp0 = pA0; sp1 = pA1; lc = LCA; ks = kr; }
                else         { sp0 = pB0; sp1 = pB1; lc = 6;   ks = kr - KA; }
                const int tap = ks >> lc;
                const int ci  = ks & ((1 << lc) - 1);
                const int dy = tap / 3 - 1, dx = tap % 3 - 1;
                { const int iy = y0 + dy, ix = x0 + dx;
                  if ((unsigned)iy < (unsigned)GH && (unsigned)ix < (unsigned)GW)
                      seg0 = *(const v8h*)(sp0 + ((((long)iy * GW + ix) << lc) + ci)); }
                { const int iy = y1 + dy, ix = x1 + dx;
                  if ((unsigned)iy < (unsigned)GH && (unsigned)ix < (unsigned)GW)
                      seg1 = *(const v8h*)(sp1 + ((((long)iy * GW + ix) << lc) + ci)); }
            }
#pragma unroll
            for (int e = 0; e < 8; ++e) { af0[run * 8 + e] = seg0[e]; af1[run * 8 + e] = seg1[e]; }
        }
        // B fragments (pre-swizzled, contiguous per lane) + 8 WMMA
        const _Float16* wb = Wswz + ((long)(kc * NTIL + wav * 4) * 32 + lane) * 16;
#pragma unroll
        for (int ni = 0; ni < 4; ++ni) {
            const v16h bf = *(const v16h*)(wb + (long)ni * 32 * 16);
            acc[ni]     = __builtin_amdgcn_wmma_f32_16x16x32_f16(false, af0, false, bf,
                                                                 (short)0, acc[ni],     false, false);
            acc[4 + ni] = __builtin_amdgcn_wmma_f32_16x16x32_f16(false, af1, false, bf,
                                                                 (short)0, acc[4 + ni], false, false);
        }
        if (kc + 1 < CHUNKS)
            __builtin_prefetch(Wswz + ((long)((kc + 1) * NTIL + wav * 4) * 32 + lane) * 16, 0, 1);
    }

    // deposit z (+bias) into LDS: local row lp -> [0,32), col n0
#pragma unroll
    for (int ni = 0; ni < 4; ++ni) {
        const int n0 = (wav * 4 + ni) * 16 + mr;
        const float bv = bias[n0];
#pragma unroll
        for (int r = 0; r < 8; ++r) {
            zbuf[(g * 8 + r) * GCO + n0]        = acc[ni][r]     + bv;
            zbuf[(16 + g * 8 + r) * GCO + n0]   = acc[4 + ni][r] + bv;
        }
    }
    __syncthreads();

    // ---------------- gates (keras order i,f,c,o) ----------------
    const int gbase = mt * 32;
#pragma unroll
    for (int ii = 0; ii < 16; ++ii) {
        const int idx = threadIdx.x + ii * 128;   // 0..2047
        const int lp = idx >> 6;
        const int ch = idx & (GF - 1);
        float zi = zbuf[lp * GCO + ch];
        float zf = zbuf[lp * GCO + GF + ch];
        float zg = zbuf[lp * GCO + 2 * GF + ch];
        float zo = zbuf[lp * GCO + 3 * GF + ch];
        float ig = fminf(fmaxf(0.2f * zi + 0.5f, 0.f), 1.f);
        float fg = fminf(fmaxf(0.2f * zf + 0.5f, 0.f), 1.f);
        float og = fminf(fmaxf(0.2f * zo + 0.5f, 0.f), 1.f);
        const long cidx = (long)(gbase + lp) * GF + ch;
        float cn = fg * cstate[cidx] + ig * tanhf(zg);
        float hv = og * tanhf(cn);
        cstate[cidx]  = cn;
        hnext16[cidx] = (_Float16)hv;
        zbuf[lp * GCO + ch] = hv;     // stash h for LayerNorm (sole reader/writer)
    }
    __syncthreads();

    // ---------------- LayerNorm(64) + LeakyReLU, sequence write ----------------
#pragma unroll
    for (int pi = 0; pi < 8; ++pi) {
        const int lp = wav * 8 + pi;
        float v0 = zbuf[lp * GCO + lane];
        float v1 = zbuf[lp * GCO + lane + 32];
        float s = v0 + v1;
#pragma unroll
        for (int off = 16; off; off >>= 1) s += __shfl_xor(s, off, 32);
        float mu = s * (1.f / 64.f);
        float d0 = v0 - mu, d1 = v1 - mu;
        float q = d0 * d0 + d1 * d1;
#pragma unroll
        for (int off = 16; off; off >>= 1) q += __shfl_xor(q, off, 32);
        float rs = rsqrtf(q * (1.f / 64.f) + 1e-3f);
        float yv0 = d0 * rs * gamma[lane]      + beta[lane];
        float yv1 = d1 * rs * gamma[lane + 32] + beta[lane + 32];
        yv0 = (yv0 >= 0.f) ? yv0 : 0.3f * yv0;
        yv1 = (yv1 >= 0.f) ? yv1 : 0.3f * yv1;
        const int gp = gbase + lp;
        const int b  = gp >> 11;
        const int rp = gp & (GHW - 1);
        const long ob = ((long)(b * T + t) * GHW + rp) * (long)GF;
        if (o16) { o16[ob + lane] = (_Float16)yv0; o16[ob + lane + 32] = (_Float16)yv1; }
        if (o32) { o32[ob + lane] = yv0;           o32[ob + lane + 32] = yv1; }
    }
}

// ---------------------------------------------------------------------------
// Attention score: score[b,t] = ba2 + sum_j tanh( r . Wa1[:,j] + ba1[j] ) * Wa2[j]
// ---------------------------------------------------------------------------
__global__ __launch_bounds__(256)
void k_score(const float* __restrict__ out, const float* __restrict__ Wa1,
             const float* __restrict__ ba1, const float* __restrict__ Wa2,
             const float* __restrict__ ba2, float* __restrict__ score, int D)
{
    __shared__ float red[256 * 16];
    const int bt = blockIdx.x;
    const float* r = out + (long)bt * D;
    float acc[16];
#pragma unroll
    for (int j = 0; j < 16; ++j) acc[j] = 0.f;
    for (int i = threadIdx.x; i < D; i += 256) {
        float rv = r[i];
        const float* wp = Wa1 + (long)i * 16;
#pragma unroll
        for (int j = 0; j < 16; ++j) acc[j] += rv * wp[j];
    }
#pragma unroll
    for (int j = 0; j < 16; ++j) red[j * 256 + threadIdx.x] = acc[j];
    __syncthreads();
    if (threadIdx.x < 16) {
        float s = 0.f;
        for (int i = 0; i < 256; ++i) s += red[threadIdx.x * 256 + i];
        red[threadIdx.x * 256] = s;
    }
    __syncthreads();
    if (threadIdx.x == 0) {
        float sc = ba2[0];
        for (int j = 0; j < 16; ++j) sc += tanhf(red[j * 256] + ba1[j]) * Wa2[j];
        score[bt] = sc;
    }
}

__global__ void k_alpha(const float* __restrict__ score, float* __restrict__ alpha,
                        int B, int T) {
    int b = blockIdx.x * blockDim.x + threadIdx.x;
    if (b >= B) return;
    float mx = -1e30f;
    for (int t = 0; t < T; ++t) mx = fmaxf(mx, score[b * T + t]);
    float s = 0.f;
    for (int t = 0; t < T; ++t) s += __expf(score[b * T + t] - mx);
    float inv = 1.f / s;
    for (int t = 0; t < T; ++t) alpha[b * T + t] = __expf(score[b * T + t] - mx) * inv;
}

__global__ void k_skip(const float* __restrict__ out, const float* __restrict__ alpha,
                       float* __restrict__ skip, int T, int D, long total) {
    long idx = (long)blockIdx.x * blockDim.x + threadIdx.x;
    if (idx >= total) return;
    int  b = (int)(idx / D);
    long i = idx - (long)b * D;
    float s = 0.f;
    for (int t = 0; t < T; ++t)
        s += alpha[b * T + t] * out[(long)(b * T + t) * D + i];
    skip[idx] = s;
}

// ---------------------------------------------------------------------------
// Launch
// ---------------------------------------------------------------------------
extern "C" void kernel_launch(void* const* d_in, const int* in_sizes, int n_in,
                              void* d_out, int out_size, void* d_ws, size_t ws_size,
                              hipStream_t stream) {
    const float* inp  = (const float*)d_in[0];
    const float* Wx1  = (const float*)d_in[1];
    const float* Wh1  = (const float*)d_in[2];
    const float* b1   = (const float*)d_in[3];
    const float* g1   = (const float*)d_in[4];
    const float* be1  = (const float*)d_in[5];
    const float* Wx2  = (const float*)d_in[6];
    const float* Wh2  = (const float*)d_in[7];
    const float* b2   = (const float*)d_in[8];
    const float* g2   = (const float*)d_in[9];
    const float* be2  = (const float*)d_in[10];
    const float* Wa1  = (const float*)d_in[11];
    const float* ba1  = (const float*)d_in[12];
    const float* Wa2  = (const float*)d_in[13];
    const float* ba2  = (const float*)d_in[14];

    const int B = 8, T = 12, Cin = 8;
    const int HW = GHW;           // 2048
    const int NP = B * HW;        // 16384
    const int D  = HW * GF;       // 131072

    const int KA1 = 9 * Cin;      // 72  (Ktot 648, 21 chunks, Kpad 672)
    const int Kpad1 = 672;
    const int KA2 = 9 * GF;       // 576 (Ktot 1152, 36 chunks)
    const int Kpad2 = 1152;

    char* ws = (char*)d_ws;
    size_t off = 0;
    auto alloc = [&](size_t bytes) -> char* {
        char* p = ws + off;
        off += (bytes + 255) & ~(size_t)255;
        return p;
    };
    _Float16* x1    = (_Float16*)alloc((size_t)B * T * HW * Cin * 2);
    _Float16* Wz1   = (_Float16*)alloc((size_t)Kpad1 * GCO * 2);
    _Float16* Wz2   = (_Float16*)alloc((size_t)Kpad2 * GCO * 2);
    float*    c1    = (float*)   alloc((size_t)NP * GF * 4);
    _Float16* h1a   = (_Float16*)alloc((size_t)NP * GF * 2);
    _Float16* h1b   = (_Float16*)alloc((size_t)NP * GF * 2);
    float*    c2    = (float*)   alloc((size_t)NP * GF * 4);
    _Float16* h2a   = (_Float16*)alloc((size_t)NP * GF * 2);
    _Float16* h2b   = (_Float16*)alloc((size_t)NP * GF * 2);
    _Float16* x2    = (_Float16*)alloc((size_t)B * T * HW * GF * 2);
    float*    score = (float*)   alloc((size_t)B * T * 4);
    float*    alpha = (float*)   alloc((size_t)B * T * 4);
    (void)ws_size; (void)in_sizes; (void)n_in; (void)out_size;

    float* out  = (float*)d_out;                       // (B,T,H,W,C)
    float* skip = out + (size_t)B * T * D;             // (B,H,W,C)

    // --- one-time per launch: f16 input + packed weights ---
    { int n = B * T * HW * Cin;
      k_f32_to_f16<<<(n + 255) / 256, 256, 0, stream>>>(inp, x1, n); }
    { int tot = Kpad1 * GCO;
      k_pack_weights<<<(tot + 255) / 256, 256, 0, stream>>>(Wx1, KA1, Wh1, 9 * GF, Kpad1, Wz1); }
    { int tot = Kpad2 * GCO;
      k_pack_weights<<<(tot + 255) / 256, 256, 0, stream>>>(Wx2, KA2, Wh2, 9 * GF, Kpad2, Wz2); }

    const int nSt = NP * GF;
    const int stepBlocks = NP / 32;    // 512 blocks of 128 threads

    // ---------------- Layer 1 ----------------
    k_zero_f32<<<(nSt + 255) / 256, 256, 0, stream>>>(c1, nSt);
    k_zero_f16<<<(nSt + 255) / 256, 256, 0, stream>>>(h1a, nSt);
    for (int t = 0; t < T; ++t) {
        _Float16* hp = (t & 1) ? h1b : h1a;
        _Float16* hn = (t & 1) ? h1a : h1b;
        k_step_fused<3, 72, 648, 21><<<stepBlocks, 128, 0, stream>>>(
            x1 + (size_t)t * HW * Cin, (long)T * HW * Cin,
            hp, Wz1, b1, c1, hn, g1, be1,
            x2, nullptr, T, t);
    }

    // ---------------- Layer 2 ----------------
    k_zero_f32<<<(nSt + 255) / 256, 256, 0, stream>>>(c2, nSt);
    k_zero_f16<<<(nSt + 255) / 256, 256, 0, stream>>>(h2a, nSt);
    for (int t = 0; t < T; ++t) {
        _Float16* hp = (t & 1) ? h2b : h2a;
        _Float16* hn = (t & 1) ? h2a : h2b;
        k_step_fused<6, 576, 1152, 36><<<stepBlocks, 128, 0, stream>>>(
            x2 + (size_t)t * HW * GF, (long)T * HW * GF,
            hp, Wz2, b2, c2, hn, g2, be2,
            nullptr, out, T, t);
    }

    // ---------------- Attention over time ----------------
    k_score<<<B * T, 256, 0, stream>>>(out, Wa1, ba1, Wa2, ba2, score, D);
    k_alpha<<<1, 32, 0, stream>>>(score, alpha, B, T);
    { long total = (long)B * D;
      k_skip<<<(int)((total + 255) / 256), 256, 0, stream>>>(out, alpha, skip, T, D, total); }
}
